// SparseCrossAttention_60816736911887
// MI455X (gfx1250) — matrix operand order repaired and checked by reading, not compiled, not run on hardware
//
#include <hip/hip_runtime.h>
#include <hip/hip_bf16.h>

typedef __attribute__((ext_vector_type(16))) _Float16 v16h;
typedef __attribute__((ext_vector_type(8)))  float    v8f;

union AF { v16h v; unsigned int u[8]; };   // 16x32 f16 A/B fragment (32 bytes)
union CF { v8f  v; float f[8]; };          // 16x16 f32 C/D fragment

static __device__ __forceinline__ v8f wmma_f16(v16h a, v16h b, v8f c) {
  // D = A(16x32 f16) * B(32x16 f16) + C(16x16 f32)
  return __builtin_amdgcn_wmma_f32_16x16x32_f16(
      /*neg_a=*/false, a, /*neg_b=*/false, b,
      /*c_mod=*/(short)0, c, /*reuse_a=*/false, /*reuse_b=*/false);
}

// Async global->LDS 16-byte copy (per lane), tracked by ASYNCcnt.
static __device__ __forceinline__ void async_b128(const void* g, void* lds) {
  asm volatile("global_load_async_to_lds_b128 %0, %1, off"
               :: "v"((unsigned)(unsigned long long)(size_t)lds),
                  "v"((unsigned long long)(size_t)g)
               : "memory");
}

// ---------------------------------------------------------------------------
// Tiled GEMM + bias:  Out[M,N] = A[M,K] * Wt^T[K,N] + bias[N]
//   Wt is pre-transposed f16 [N][K] so B tiles are raw row copies that land
//   directly in the WMMA B-fragment LDS layout -> async global->LDS DMA.
//   A: f16 -> async DMA; f32 -> load/convert/ds_store path.
// Double-buffered K tiles: issue async for kt+1, s_wait_asynccnt, compute kt.
// Block tile 128x128, 8 waves (2x4), wave tile 64x32, 8 WMMA per K step.
// ---------------------------------------------------------------------------
template <typename InT, typename OutT>
__global__ __launch_bounds__(256)
void gemm_bias_kernel(const InT* __restrict__ A, const _Float16* __restrict__ Wt,
                      const float* __restrict__ bias, OutT* __restrict__ Out,
                      int M, int K, int N) {
  constexpr int BK = 32;
  constexpr int LDSTR = 40;                 // 80B row stride: 16B aligned
  constexpr bool ASYNC_A = (sizeof(InT) == 2);
  __shared__ _Float16 As[2][128][LDSTR];
  __shared__ _Float16 Bs[2][128][LDSTR];

  const int tid  = threadIdx.x;
  const int lane = tid & 31;
  const int wave = tid >> 5;
  const int wm   = wave >> 2;               // 0..1  (M sub-block of 64)
  const int wn   = wave & 3;                // 0..3  (N sub-block of 32)
  const int m0   = blockIdx.x * 128;
  const int n0   = blockIdx.y * 128;

  CF acc[4][2];
  #pragma unroll
  for (int i = 0; i < 4; ++i)
    #pragma unroll
    for (int j = 0; j < 2; ++j)
      #pragma unroll
      for (int e = 0; e < 8; ++e) acc[i][j].f[e] = 0.0f;

  const int kTiles = K / BK;                // 8
  const int hl = lane >> 4;
  const int kb = hl * 16;                   // unused (B fragment uses ko)
  const int ko = hl * 8;                    // A-fragment K base
  const int ln = lane & 15;

  auto stage = [&](int kt, int p) {
    // B tile: 128 rows x 64B from Wt -> 2 async b128 chunks per thread
    #pragma unroll
    for (int c = 0; c < 2; ++c) {
      int chunk = c * 256 + tid;
      int row = chunk >> 2, seg = chunk & 3;
      async_b128(Wt + (size_t)(n0 + row) * K + kt * BK + seg * 8,
                 &Bs[p][row][seg * 8]);
    }
    if constexpr (ASYNC_A) {
      #pragma unroll
      for (int c = 0; c < 2; ++c) {
        int chunk = c * 256 + tid;
        int row = chunk >> 2, seg = chunk & 3;
        async_b128(A + (size_t)(m0 + row) * K + kt * BK + seg * 8,
                   &As[p][row][seg * 8]);
      }
    } else {
      #pragma unroll
      for (int pass = 0; pass < 4; ++pass) {
        int flat = (pass * 256 + tid) * 4;
        int col = flat & 31, row = flat >> 5;
        const InT* src = A + (size_t)(m0 + row) * K + kt * BK + col;
        #pragma unroll
        for (int i = 0; i < 4; ++i) As[p][row][col + i] = (_Float16)src[i];
      }
    }
  };

  stage(0, 0);                              // prologue: tile 0 in flight

  for (int kt = 0; kt < kTiles; ++kt) {
    const int p = kt & 1;
    if (kt + 1 < kTiles) {
      stage(kt + 1, p ^ 1);                 // keep next tile's DMA in flight
      if constexpr (ASYNC_A)
        asm volatile("s_wait_asynccnt 0x4" ::: "memory");
      else
        asm volatile("s_wait_asynccnt 0x2" ::: "memory");
    } else {
      asm volatile("s_wait_asynccnt 0x0" ::: "memory");
    }
    __syncthreads();                        // tile kt visible to all waves

    AF bf[2];
    #pragma unroll
    for (int ni = 0; ni < 2; ++ni) {
      int n = wn * 32 + ni * 16 + ln;
      #pragma unroll
      for (int q = 0; q < 8; ++q)
        bf[ni].u[q] = *(const unsigned int*)&Bs[p][n][ko + 2 * q + (q >> 2) * 8];
    }
    #pragma unroll
    for (int mi = 0; mi < 4; ++mi) {
      AF af;
      int m = wm * 64 + mi * 16 + ln;
      #pragma unroll
      for (int q = 0; q < 4; ++q) {
        af.u[q]     = *(const unsigned int*)&As[p][m][ko + 2 * q];
        af.u[4 + q] = *(const unsigned int*)&As[p][m][16 + ko + 2 * q];
      }
      #pragma unroll
      for (int ni = 0; ni < 2; ++ni)
        acc[mi][ni].v = wmma_f16(af.v, bf[ni].v, acc[mi][ni].v);
    }
    __syncthreads();                        // done reading buf p (rewritten at kt+2)
  }

  // epilogue: bias + store (C/D layout: M = j + 8*hl, N = lane&15)
  #pragma unroll
  for (int ni = 0; ni < 2; ++ni) {
    int n = n0 + wn * 32 + ni * 16 + ln;
    float b = bias[n];
    #pragma unroll
    for (int mi = 0; mi < 4; ++mi) {
      #pragma unroll
      for (int j = 0; j < 8; ++j) {
        int m = m0 + wm * 64 + mi * 16 + j + hl * 8;
        Out[(size_t)m * N + n] = (OutT)(acc[mi][ni].f[j] + b);
      }
    }
  }
}

// ---------------------------------------------------------------------------
// Weight transpose+convert: Wt[n][k] = (f16) W[k][n]   (256x256)
// ---------------------------------------------------------------------------
__global__ __launch_bounds__(256)
void wcvt_kernel(const float* __restrict__ W, _Float16* __restrict__ Wt) {
  int k = blockIdx.x;
  int n = threadIdx.x;
  Wt[(size_t)n * 256 + k] = (_Float16)W[(size_t)k * 256 + n];
}

// ---------------------------------------------------------------------------
// Windowed attention: one workgroup = (window, 4 heads), 4 waves, each wave
// owns one head end-to-end (wave-private LDS -> no barriers).
// Per wave: S = Q K^T (16 WMMA), shuffle-reduced softmax, P V (16 WMMA).
// ---------------------------------------------------------------------------
__global__ __launch_bounds__(128)
void win_attn_kernel(const _Float16* __restrict__ Qp,
                     const _Float16* __restrict__ Kp,
                     const _Float16* __restrict__ Vp,
                     _Float16* __restrict__ Op) {
  __shared__ _Float16 Ps[4][64][72];   // softmax probs, per wave
  __shared__ _Float16 Vs[4][32][72];   // V transposed [d][token], per wave

  const int tid  = threadIdx.x;
  const int lane = tid & 31;
  const int w    = tid >> 5;
  const int h    = blockIdx.z * 4 + w;          // head 0..7
  const int wIdx = blockIdx.x;                  // window 0..255
  const int wh   = wIdx >> 4, ww = wIdx & 15;
  const size_t batchBase = (size_t)blockIdx.y * 16384;
  const int ch0 = h * 32;
  const int hl  = lane >> 4;
  const int ln  = lane & 15;
  const int kb16 = hl * 16;
  const int ko8  = hl * 8;

  auto rowOf = [&](int t) -> size_t {           // window token -> global row
    int r = t >> 3, c = t & 7;
    return batchBase + (size_t)((wh * 8 + r) * 128 + ww * 8 + c);
  };

  // stage V^T: Vs[w][d][token]  (lane = d, coalesced global reads)
  for (int t = 0; t < 64; ++t)
    Vs[w][lane][t] = Vp[rowOf(t) * 256 + ch0 + lane];

  // K fragments (B-matrix: K-dim = d, N = key token); table k order per lane
  AF kf[4];
  #pragma unroll
  for (int ni = 0; ni < 4; ++ni) {
    int t = ni * 16 + ln;
    const _Float16* src = Kp + rowOf(t) * 256 + ch0;
    #pragma unroll
    for (int q = 0; q < 8; ++q)
      kf[ni].u[q] = *(const unsigned int*)&src[ko8 + 2 * q + (q >> 2) * 8];
  }

  // S = Q K^T  (M=query, N=key, K=32=head_dim)
  CF sacc[4][4];
  #pragma unroll
  for (int mi = 0; mi < 4; ++mi)
    #pragma unroll
    for (int ni = 0; ni < 4; ++ni)
      #pragma unroll
      for (int e = 0; e < 8; ++e) sacc[mi][ni].f[e] = 0.0f;

  #pragma unroll
  for (int mi = 0; mi < 4; ++mi) {
    AF qf;
    int t = mi * 16 + ln;
    const _Float16* src = Qp + rowOf(t) * 256 + ch0;
    #pragma unroll
    for (int q = 0; q < 4; ++q) {
      qf.u[q]     = *(const unsigned int*)&src[ko8 + 2 * q];
      qf.u[4 + q] = *(const unsigned int*)&src[16 + ko8 + 2 * q];
    }
    #pragma unroll
    for (int ni = 0; ni < 4; ++ni)
      sacc[mi][ni].v = wmma_f16(qf.v, kf[ni].v, sacc[mi][ni].v);
  }

  // softmax per row (64 keys over 4 ni-tiles x 16 lanes of one half-wave)
  const float scale = 0.17677669529663687f;   // 1/sqrt(32)
  #pragma unroll
  for (int mi = 0; mi < 4; ++mi) {
    #pragma unroll
    for (int j = 0; j < 8; ++j) {
      float v0 = sacc[mi][0].f[j] * scale;
      float v1 = sacc[mi][1].f[j] * scale;
      float v2 = sacc[mi][2].f[j] * scale;
      float v3 = sacc[mi][3].f[j] * scale;
      float mx = fmaxf(fmaxf(v0, v1), fmaxf(v2, v3));
      #pragma unroll
      for (int off = 1; off < 16; off <<= 1)
        mx = fmaxf(mx, __shfl_xor(mx, off, 32));   // stays inside 16-lane half
      float e0 = exp2f((v0 - mx) * 1.44269504f);
      float e1 = exp2f((v1 - mx) * 1.44269504f);
      float e2 = exp2f((v2 - mx) * 1.44269504f);
      float e3 = exp2f((v3 - mx) * 1.44269504f);
      float s = e0 + e1 + e2 + e3;
      #pragma unroll
      for (int off = 1; off < 16; off <<= 1)
        s += __shfl_xor(s, off, 32);
      float inv = 1.0f / s;
      int m = mi * 16 + j + hl * 8;
      Ps[w][m][ 0 + ln] = (_Float16)(e0 * inv);
      Ps[w][m][16 + ln] = (_Float16)(e1 * inv);
      Ps[w][m][32 + ln] = (_Float16)(e2 * inv);
      Ps[w][m][48 + ln] = (_Float16)(e3 * inv);
    }
  }

  // O = P V  (M=query, K=key(64 -> 2 steps), N=d(32 -> 2 tiles))
  CF oacc[4][2];
  #pragma unroll
  for (int mi = 0; mi < 4; ++mi)
    #pragma unroll
    for (int ni = 0; ni < 2; ++ni)
      #pragma unroll
      for (int e = 0; e < 8; ++e) oacc[mi][ni].f[e] = 0.0f;

  #pragma unroll
  for (int ks = 0; ks < 2; ++ks) {
    AF vf[2];
    #pragma unroll
    for (int ni = 0; ni < 2; ++ni) {
      int d = ni * 16 + ln;
      #pragma unroll
      for (int q = 0; q < 8; ++q)
        vf[ni].u[q] = *(const unsigned int*)&Vs[w][d][ks * 32 + ko8 + 2 * q + (q >> 2) * 8];
    }
    #pragma unroll
    for (int mi = 0; mi < 4; ++mi) {
      AF pf;
      int m = mi * 16 + ln;
      #pragma unroll
      for (int q = 0; q < 4; ++q) {
        pf.u[q]     = *(const unsigned int*)&Ps[w][m][ks * 32 + ko8 + 2 * q];
        pf.u[4 + q] = *(const unsigned int*)&Ps[w][m][ks * 32 + 16 + ko8 + 2 * q];
      }
      #pragma unroll
      for (int ni = 0; ni < 2; ++ni)
        oacc[mi][ni].v = wmma_f16(pf.v, vf[ni].v, oacc[mi][ni].v);
    }
  }

  // window-reverse store into (B, N, C) layout, f16 for final projection
  #pragma unroll
  for (int ni = 0; ni < 2; ++ni) {
    int d = ni * 16 + ln;
    #pragma unroll
    for (int mi = 0; mi < 4; ++mi) {
      #pragma unroll
      for (int j = 0; j < 8; ++j) {
        int t = mi * 16 + j + hl * 8;
        Op[rowOf(t) * 256 + ch0 + d] = (_Float16)oacc[mi][ni].f[j];
      }
    }
  }
}

// ---------------------------------------------------------------------------
extern "C" void kernel_launch(void* const* d_in, const int* in_sizes, int n_in,
                              void* d_out, int out_size, void* d_ws, size_t ws_size,
                              hipStream_t stream) {
  const float* q  = (const float*)d_in[0];
  const float* kv = (const float*)d_in[1];
  const float* Wq = (const float*)d_in[2];
  const float* bq = (const float*)d_in[3];
  const float* Wk = (const float*)d_in[4];
  const float* bk = (const float*)d_in[5];
  const float* Wv = (const float*)d_in[6];
  const float* bv = (const float*)d_in[7];
  const float* Wo = (const float*)d_in[8];
  const float* bo = (const float*)d_in[9];
  float* out = (float*)d_out;

  const int Mrows = 4 * 16384;   // B*N
  const int C = 256;
  const size_t planeBytes = (size_t)Mrows * C * sizeof(_Float16);  // 32 MB
  const size_t wtBytes = (size_t)C * C * sizeof(_Float16);         // 128 KB
  char* ws = (char*)d_ws;
  _Float16* qp  = (_Float16*)(ws + 0 * planeBytes);
  _Float16* kp  = (_Float16*)(ws + 1 * planeBytes);
  _Float16* vp  = (_Float16*)(ws + 2 * planeBytes);
  _Float16* ao  = (_Float16*)(ws + 3 * planeBytes);
  _Float16* WqT = (_Float16*)(ws + 4 * planeBytes);
  _Float16* WkT = (_Float16*)((char*)WqT + wtBytes);
  _Float16* WvT = (_Float16*)((char*)WkT + wtBytes);
  _Float16* WoT = (_Float16*)((char*)WvT + wtBytes);

  // one-time weight transpose+convert to f16 (async-DMA friendly layout)
  wcvt_kernel<<<C, C, 0, stream>>>(Wq, WqT);
  wcvt_kernel<<<C, C, 0, stream>>>(Wk, WkT);
  wcvt_kernel<<<C, C, 0, stream>>>(Wv, WvT);
  wcvt_kernel<<<C, C, 0, stream>>>(Wo, WoT);

  dim3 gemmGrid(Mrows / 128, C / 128, 1);   // 512 x 2

  gemm_bias_kernel<float, _Float16><<<gemmGrid, 256, 0, stream>>>(q,  WqT, bq, qp, Mrows, C, C);
  gemm_bias_kernel<float, _Float16><<<gemmGrid, 256, 0, stream>>>(kv, WkT, bk, kp, Mrows, C, C);
  gemm_bias_kernel<float, _Float16><<<gemmGrid, 256, 0, stream>>>(kv, WvT, bv, vp, Mrows, C, C);

  // 256 windows x 4 batches x 2 head-groups; 128 threads = 4 waves = 4 heads
  win_attn_kernel<<<dim3(256, 4, 2), 128, 0, stream>>>(qp, kp, vp, ao);

  gemm_bias_kernel<_Float16, float><<<gemmGrid, 256, 0, stream>>>(ao, WoT, bo, out, Mrows, C, C);
}
